// SCAttention_75814762709306
// MI455X (gfx1250) — matrix operand order repaired and checked by reading, not compiled, not run on hardware
//
#include <hip/hip_runtime.h>

// Problem dims (fixed by the reference)
#define B_ 16
#define P_ 2048
#define Q_ 2048
#define D_ 1024
#define H_ 1024

typedef __bf16 bf16_t;
typedef __attribute__((ext_vector_type(16))) __bf16 bf16x16;
typedef __attribute__((ext_vector_type(8)))  __bf16 bf16x8;
typedef __attribute__((ext_vector_type(8)))  float  f32x8;
typedef __attribute__((ext_vector_type(4)))  unsigned int u32x4;
typedef __attribute__((ext_vector_type(8)))  int i32x8;
typedef __attribute__((ext_vector_type(4)))  int i32x4;

// ---------------------------------------------------------------------------
// Tensor Data Mover: DMA a 2D tile (rows x rowElems, row-major) from global
// memory into LDS. D# built per CDNA5 ISA ch.8. LDS padding: +16B per 1KB
// stored (pad_interval=256 DWORDs -> code 7, pad_amount=4 DWORDs -> code 3)
// to de-stripe the 2KB/4KB row-stride bank pattern while keeping every
// fragment chunk 16B-aligned.
// ---------------------------------------------------------------------------
__device__ __forceinline__ unsigned lds_offset(const void* p) {
  // Generic LDS addresses carry the byte offset in the low 32 bits (ISA 10.2)
  return (unsigned)(unsigned long long)p;
}

__device__ __forceinline__ void tdm_load_2d(const void* gsrc, unsigned ldsOff,
                                            unsigned rowElems, unsigned rows,
                                            unsigned strideElems, int dataSizeCode) {
  unsigned long long ga = (unsigned long long)gsrc;
  u32x4 g0;
  g0[0] = 1u;  // count=1 valid descriptor, user mode
  g0[1] = ldsOff;
  g0[2] = (unsigned)ga;
  g0[3] = (unsigned)((ga >> 32) & 0x01FFFFFFu) | (2u << 30);  // type=2 (image)
  i32x8 g1;
  g1[0] = (int)(((unsigned)dataSizeCode << 16) | (1u << 20) | (7u << 22) |
                (3u << 25));                                   // pad: 16B / 1KB
  g1[1] = (int)((rowElems & 0xFFFFu) << 16);                   // tensor_dim0 lo16
  g1[2] = (int)(((rowElems >> 16) & 0xFFFFu) | ((rows & 0xFFFFu) << 16));
  g1[3] = (int)(((rows >> 16) & 0xFFFFu) | ((rowElems & 0xFFFFu) << 16)); // tile_dim0
  g1[4] = (int)(rows & 0xFFFFu);                               // tile_dim1, tile_dim2=0
  g1[5] = (int)strideElems;                                    // dim0 stride lo32
  g1[6] = 0;
  g1[7] = 0;
  i32x4 z4 = {0, 0, 0, 0};
#if __has_builtin(__builtin_amdgcn_tensor_load_to_lds)
#if defined(__clang_major__) && (__clang_major__ >= 23)
  i32x8 z8 = {0, 0, 0, 0, 0, 0, 0, 0};
  __builtin_amdgcn_tensor_load_to_lds(g0, g1, z4, z4, z8, 0);
#else
  __builtin_amdgcn_tensor_load_to_lds(g0, g1, z4, z4, 0);
#endif
#endif
}

__device__ __forceinline__ void tdm_wait() {
#if __has_builtin(__builtin_amdgcn_s_wait_tensorcnt)
  __builtin_amdgcn_s_wait_tensorcnt(0);
#endif
}

// LDS pitch (bytes) of a padded row of rowBytes
#define LDS_PITCH(rowBytes) ((rowBytes) + (((rowBytes) >> 10) << 4))

// ---------------------------------------------------------------------------
// Fragment loaders matching CDNA5 wave32 WMMA VGPR layouts (ISA 7.12.2).
// A (16x32 bf16): lanes 0-15: M=lane, K={0..7,16..23}; lanes 16-31: M=lane-16,
// K={8..15,24..31}. B (32x16): lane n<16: col n K={0..7,16..23}; lane n+16: K={8..15,24..31}.
// ---------------------------------------------------------------------------
__device__ __forceinline__ bf16x16 load_a_lds_bf16(const char* smem, int pitch,
                                                   int mBase, int k0, int lane) {
  const int half = lane >> 4;
  const int m = mBase + (lane & 15);
  const int kb = k0 * 2;
  const char* p = smem + m * pitch + kb + ((kb >> 10) << 4) + half * 16;
  bf16x8 lo = *(const bf16x8*)p;
  bf16x8 hi = *(const bf16x8*)(p + 32);
  bf16x16 f;
#pragma unroll
  for (int i = 0; i < 8; ++i) { f[i] = lo[i]; f[i + 8] = hi[i]; }
  return f;
}

__device__ __forceinline__ bf16x16 load_a_lds_f32(const char* smem, int pitch,
                                                  int mBase, int k0, int lane) {
  const int half = lane >> 4;
  const int m = mBase + (lane & 15);
  const int kb = k0 * 4;
  const char* p = smem + m * pitch + kb + ((kb >> 10) << 4) + half * 32;
  f32x8 lo = *(const f32x8*)p;
  f32x8 hi = *(const f32x8*)(p + 64);
  bf16x16 f;
#pragma unroll
  for (int i = 0; i < 8; ++i) { f[i] = (bf16_t)lo[i]; f[i + 8] = (bf16_t)hi[i]; }
  return f;
}

// BT stored [n][k] (K contiguous): fragment for B[k0..k0+31][n0..n0+15]
__device__ __forceinline__ bf16x16 load_b(const bf16_t* __restrict__ BT, int ldb,
                                          int n0, int k0, int lane) {
  const int n = lane & 15;
  const int kb = (lane >> 4) * 8;
  const bf16_t* p = BT + (size_t)(n0 + n) * ldb + k0 + kb;
  bf16x8 lo = *(const bf16x8*)p;
  bf16x8 hi = *(const bf16x8*)(p + 16);
  bf16x16 f;
#pragma unroll
  for (int i = 0; i < 8; ++i) { f[i] = lo[i]; f[i + 8] = hi[i]; }
  return f;
}

__device__ __forceinline__ f32x8 wmma_bf16(bf16x16 a, bf16x16 b, f32x8 c) {
  return __builtin_amdgcn_wmma_f32_16x16x32_bf16(false, a, false, b, (short)0, c,
                                                 false, false);
}

// ---------------------------------------------------------------------------
// Kernel 0: transpose + f32->bf16 convert (small weight matrices)
// ---------------------------------------------------------------------------
__global__ void k_transpose_cvt(const float* __restrict__ src, bf16_t* __restrict__ dst,
                                int R, int C) {
  int idx = blockIdx.x * blockDim.x + threadIdx.x;
  if (idx >= R * C) return;
  int r = idx / C, c = idx - r * C;
  dst[(size_t)c * R + r] = (bf16_t)src[idx];
}

// ---------------------------------------------------------------------------
// Kernel 1: Y[M,N](bf16) = f32 X[M,K] @ W (+bias); W staged as WT[n][k] bf16.
// 32 M-rows per block: one TDM DMA stages the whole f32 A stripe in LDS;
// each wave computes 2 M-tiles x 4 N-tiles, so every B fragment feeds 2
// WMMAs (halves global B traffic per FLOP). B fragments ping-pong between
// two register buffers (K unrolled x2) - no rotation moves.
// ---------------------------------------------------------------------------
__global__ void __launch_bounds__(128)
k_gemm_xw(const float* __restrict__ X, const bf16_t* __restrict__ WT,
          const float* __restrict__ bias, bf16_t* __restrict__ Y,
          bf16_t* __restrict__ YT, int K, int N, int rowsPerBatch) {
  __shared__ char smem[32 * LDS_PITCH(D_ * 4)];  // 32 rows x padded 4KB (~130KB)
  const int lane = threadIdx.x & 31;
  const int w = threadIdx.x >> 5;
  const int half = lane >> 4;
  const int m0 = blockIdx.x * 32;
  const int nbase = blockIdx.y * 256 + w * 64;
  const int pitch = LDS_PITCH(K * 4);

  if (threadIdx.x == 0) {
    tdm_load_2d(X + (size_t)m0 * K, lds_offset(smem), K, 32, K, /*f32*/ 2);
    tdm_wait();
  }
  __syncthreads();

  f32x8 acc[2][4];
#pragma unroll
  for (int mt = 0; mt < 2; ++mt)
#pragma unroll
    for (int t = 0; t < 4; ++t)
#pragma unroll
      for (int r = 0; r < 8; ++r) acc[mt][t][r] = 0.f;

  bf16x16 b0[4], b1[4];
#pragma unroll
  for (int t = 0; t < 4; ++t) b0[t] = load_b(WT, K, nbase + t * 16, 0, lane);

#pragma unroll 1
  for (int k0 = 0; k0 < K; k0 += 64) {  // K % 64 == 0 for all uses
#pragma unroll
    for (int t = 0; t < 4; ++t) b1[t] = load_b(WT, K, nbase + t * 16, k0 + 32, lane);
    {
      bf16x16 a0 = load_a_lds_f32(smem, pitch, 0, k0, lane);
      bf16x16 a1 = load_a_lds_f32(smem, pitch, 16, k0, lane);
#pragma unroll
      for (int t = 0; t < 4; ++t) {
        acc[0][t] = wmma_bf16(a0, b0[t], acc[0][t]);
        acc[1][t] = wmma_bf16(a1, b0[t], acc[1][t]);
      }
    }
    const int kn = (k0 + 64 < K) ? (k0 + 64) : k0;  // harmless reload on last iter
#pragma unroll
    for (int t = 0; t < 4; ++t) b0[t] = load_b(WT, K, nbase + t * 16, kn, lane);
    {
      bf16x16 a0 = load_a_lds_f32(smem, pitch, 0, k0 + 32, lane);
      bf16x16 a1 = load_a_lds_f32(smem, pitch, 16, k0 + 32, lane);
#pragma unroll
      for (int t = 0; t < 4; ++t) {
        acc[0][t] = wmma_bf16(a0, b1[t], acc[0][t]);
        acc[1][t] = wmma_bf16(a1, b1[t], acc[1][t]);
      }
    }
  }

#pragma unroll
  for (int mt = 0; mt < 2; ++mt)
#pragma unroll
    for (int t = 0; t < 4; ++t) {
      const int col = nbase + t * 16 + (lane & 15);
      const float bv = bias[col];
#pragma unroll
      for (int r = 0; r < 8; ++r) {
        const int row = m0 + mt * 16 + half * 8 + r;
        const float v = acc[mt][t][r] + bv;
        Y[(size_t)row * N + col] = (bf16_t)v;
        if (YT) {
          const int bt = row / rowsPerBatch;
          const int q = row - bt * rowsPerBatch;
          YT[((size_t)bt * N + col) * rowsPerBatch + q] = (bf16_t)v;
        }
      }
    }
}

// ---------------------------------------------------------------------------
// Kernel 2: fused scores + masked softmax -> alpha (bf16).
// One block (8 waves) owns a 16 x 2048 stripe of scores[b, p0:p0+16, :].
// The shared Wp stripe (16 x H bf16) is TDM-staged in LDS; Wq streams
// through L2 (4MB/batch << 192MB). Faithful renorm:
// alpha = m*e / (sum(m*e) + 1e-13*sum(e)).
// ---------------------------------------------------------------------------
__global__ void __launch_bounds__(256)
k_scores_softmax(const bf16_t* __restrict__ Wp, const bf16_t* __restrict__ Wq,
                 const int* __restrict__ qmask, bf16_t* __restrict__ alpha) {
  __shared__ char smem[16 * LDS_PITCH(H_ * 2)];  // 16 rows x padded 2KB
  const int lane = threadIdx.x & 31;
  const int w = threadIdx.x >> 5;
  const int half = lane >> 4;
  const int tilesPerRow = P_ / 16;  // 128
  const int b = blockIdx.x / tilesPerRow;
  const int p0 = (blockIdx.x - b * tilesPerRow) * 16;
  const bf16_t* Wpb = Wp + (size_t)b * P_ * H_;
  const bf16_t* Wqb = Wq + (size_t)b * Q_ * H_;
  const int nbase = w * 256;
  const int pitch = LDS_PITCH(H_ * 2);

  if (threadIdx.x == 0) {
    tdm_load_2d(Wpb + (size_t)p0 * H_, lds_offset(smem), H_, 16, H_, /*bf16*/ 1);
    tdm_wait();
  }
  __syncthreads();

  f32x8 acc[16];
#pragma unroll
  for (int t = 0; t < 16; ++t)
#pragma unroll
    for (int r = 0; r < 8; ++r) acc[t][r] = 0.f;

#pragma unroll 1
  for (int k0 = 0; k0 < H_; k0 += 32) {
    bf16x16 a = load_a_lds_bf16(smem, pitch, 0, k0, lane);
#pragma unroll
    for (int t = 0; t < 16; ++t) {
      bf16x16 bb = load_b(Wqb, H_, nbase + t * 16, k0, lane);
      acc[t] = wmma_bf16(a, bb, acc[t]);
    }
  }

  // Column masks; s *= mask (reference: softmax(scores*mask))
  float mk[16];
#pragma unroll
  for (int t = 0; t < 16; ++t)
    mk[t] = (float)qmask[b * Q_ + nbase + t * 16 + (lane & 15)];
#pragma unroll
  for (int t = 0; t < 16; ++t)
#pragma unroll
    for (int r = 0; r < 8; ++r) acc[t][r] *= mk[t];

  // ---- row max over all Q ----
  float lmax[8];
#pragma unroll
  for (int r = 0; r < 8; ++r) {
    float v = -3.402823466e38f;
#pragma unroll
    for (int t = 0; t < 16; ++t) v = fmaxf(v, acc[t][r]);
    lmax[r] = v;
  }
#pragma unroll
  for (int off = 1; off <= 8; off <<= 1)
#pragma unroll
    for (int r = 0; r < 8; ++r) lmax[r] = fmaxf(lmax[r], __shfl_xor(lmax[r], off, 32));

  __shared__ float redM[8][16];
  if ((lane & 15) == 0)
#pragma unroll
    for (int r = 0; r < 8; ++r) redM[w][half * 8 + r] = lmax[r];
  __syncthreads();
  float rmax[8];
#pragma unroll
  for (int r = 0; r < 8; ++r) {
    const int m = half * 8 + r;
    float v = redM[0][m];
#pragma unroll
    for (int w2 = 1; w2 < 8; ++w2) v = fmaxf(v, redM[w2][m]);
    rmax[r] = v;
  }

  // ---- exp + (all, masked) sums ----
  float s1[8], s2[8];
#pragma unroll
  for (int r = 0; r < 8; ++r) { s1[r] = 0.f; s2[r] = 0.f; }
#pragma unroll
  for (int t = 0; t < 16; ++t)
#pragma unroll
    for (int r = 0; r < 8; ++r) {
      float e = __expf(acc[t][r] - rmax[r]);
      s1[r] += e;
      e *= mk[t];
      s2[r] += e;
      acc[t][r] = e;
    }
#pragma unroll
  for (int off = 1; off <= 8; off <<= 1)
#pragma unroll
    for (int r = 0; r < 8; ++r) {
      s1[r] += __shfl_xor(s1[r], off, 32);
      s2[r] += __shfl_xor(s2[r], off, 32);
    }
  __shared__ float redS1[8][16];
  __shared__ float redS2[8][16];
  if ((lane & 15) == 0)
#pragma unroll
    for (int r = 0; r < 8; ++r) {
      redS1[w][half * 8 + r] = s1[r];
      redS2[w][half * 8 + r] = s2[r];
    }
  __syncthreads();
  float rinv[8];
#pragma unroll
  for (int r = 0; r < 8; ++r) {
    const int m = half * 8 + r;
    float a1 = 0.f, a2 = 0.f;
#pragma unroll
    for (int w2 = 0; w2 < 8; ++w2) { a1 += redS1[w2][m]; a2 += redS2[w2][m]; }
    rinv[r] = 1.0f / (a2 + 1e-13f * a1);
  }

  bf16_t* ab = alpha + (size_t)b * P_ * Q_;
#pragma unroll
  for (int t = 0; t < 16; ++t) {
    const int col = nbase + t * 16 + (lane & 15);
#pragma unroll
    for (int r = 0; r < 8; ++r) {
      const int row = p0 + half * 8 + r;
      ab[(size_t)row * Q_ + col] = (bf16_t)(acc[t][r] * rinv[r]);
    }
  }
}

// ---------------------------------------------------------------------------
// Kernel 3/4: Y = act(A(bf16)[M,K] @ B (+bias)); B staged as BT[n][k] bf16,
// optionally per-batch (btStride elems). 32 M-rows per block via one TDM DMA;
// each wave does 2 M-tiles x 4 N-tiles with ping-pong B buffers.
// Output bf16 (Yb) and/or f32 (Yf).
// ---------------------------------------------------------------------------
__global__ void __launch_bounds__(128)
k_gemm_av(const bf16_t* __restrict__ A, const bf16_t* __restrict__ BT,
          const float* __restrict__ bias, bf16_t* __restrict__ Yb,
          float* __restrict__ Yf, int K, int N, long btStride, int rowsPerBatch,
          int relu) {
  __shared__ char smem[32 * LDS_PITCH(Q_ * 2)];  // worst case K=2048 (~130KB)
  const int lane = threadIdx.x & 31;
  const int w = threadIdx.x >> 5;
  const int half = lane >> 4;
  const int m0 = blockIdx.x * 32;
  const int nbase = blockIdx.y * 256 + w * 64;
  const int batch = m0 / rowsPerBatch;
  const bf16_t* BTb = BT + (size_t)batch * btStride;
  const int pitch = LDS_PITCH(K * 2);

  if (threadIdx.x == 0) {
    tdm_load_2d(A + (size_t)m0 * K, lds_offset(smem), K, 32, K, /*bf16*/ 1);
    tdm_wait();
  }
  __syncthreads();

  f32x8 acc[2][4];
#pragma unroll
  for (int mt = 0; mt < 2; ++mt)
#pragma unroll
    for (int t = 0; t < 4; ++t)
#pragma unroll
      for (int r = 0; r < 8; ++r) acc[mt][t][r] = 0.f;

  bf16x16 b0[4], b1[4];
#pragma unroll
  for (int t = 0; t < 4; ++t) b0[t] = load_b(BTb, K, nbase + t * 16, 0, lane);

#pragma unroll 1
  for (int k0 = 0; k0 < K; k0 += 64) {  // K % 64 == 0 for all uses
#pragma unroll
    for (int t = 0; t < 4; ++t) b1[t] = load_b(BTb, K, nbase + t * 16, k0 + 32, lane);
    {
      bf16x16 a0 = load_a_lds_bf16(smem, pitch, 0, k0, lane);
      bf16x16 a1 = load_a_lds_bf16(smem, pitch, 16, k0, lane);
#pragma unroll
      for (int t = 0; t < 4; ++t) {
        acc[0][t] = wmma_bf16(a0, b0[t], acc[0][t]);
        acc[1][t] = wmma_bf16(a1, b0[t], acc[1][t]);
      }
    }
    const int kn = (k0 + 64 < K) ? (k0 + 64) : k0;
#pragma unroll
    for (int t = 0; t < 4; ++t) b0[t] = load_b(BTb, K, nbase + t * 16, kn, lane);
    {
      bf16x16 a0 = load_a_lds_bf16(smem, pitch, 0, k0 + 32, lane);
      bf16x16 a1 = load_a_lds_bf16(smem, pitch, 16, k0 + 32, lane);
#pragma unroll
      for (int t = 0; t < 4; ++t) {
        acc[0][t] = wmma_bf16(a0, b1[t], acc[0][t]);
        acc[1][t] = wmma_bf16(a1, b1[t], acc[1][t]);
      }
    }
  }

#pragma unroll
  for (int mt = 0; mt < 2; ++mt)
#pragma unroll
    for (int t = 0; t < 4; ++t) {
      const int col = nbase + t * 16 + (lane & 15);
      const float bv = bias ? bias[col] : 0.f;
#pragma unroll
      for (int r = 0; r < 8; ++r) {
        const int row = m0 + mt * 16 + half * 8 + r;
        float v = acc[mt][t][r] + bv;
        if (relu) v = fmaxf(v, 0.f);
        if (Yb) Yb[(size_t)row * N + col] = (bf16_t)v;
        if (Yf) Yf[(size_t)row * N + col] = v;
      }
    }
}

// ---------------------------------------------------------------------------
// Host-side orchestration
// ---------------------------------------------------------------------------
extern "C" void kernel_launch(void* const* d_in, const int* in_sizes, int n_in,
                              void* d_out, int out_size, void* d_ws, size_t ws_size,
                              hipStream_t stream) {
  (void)in_sizes; (void)n_in; (void)out_size; (void)ws_size;
  const float* passage  = (const float*)d_in[0];  // [B,P,D]
  const float* question = (const float*)d_in[1];  // [B,Q,D]
  const int*   q_mask   = (const int*)d_in[2];    // [B,Q]
  const float* W_w      = (const float*)d_in[3];  // [D,H]
  const float* W_b      = (const float*)d_in[4];  // [H]
  const float* map_w    = (const float*)d_in[5];  // [H,H]
  const float* map_b    = (const float*)d_in[6];  // [H]
  float* out = (float*)d_out;                     // [B,P,H] f32

  char* ws = (char*)d_ws;
  size_t off = 0;
  bf16_t* WwT  = (bf16_t*)(ws + off); off += (size_t)H_ * D_ * 2;      //  2 MB [h][d]
  bf16_t* mapT = (bf16_t*)(ws + off); off += (size_t)H_ * H_ * 2;      //  2 MB [ho][hi]
  bf16_t* Wp   = (bf16_t*)(ws + off); off += (size_t)B_ * P_ * H_ * 2; // 64 MB
  bf16_t* Wq   = (bf16_t*)(ws + off); off += (size_t)B_ * Q_ * H_ * 2; // 64 MB
  bf16_t* WqT  = (bf16_t*)(ws + off); off += (size_t)B_ * H_ * Q_ * 2; // 64 MB
  bf16_t* alp  = (bf16_t*)(ws + off); off += (size_t)B_ * P_ * Q_ * 2; // 128 MB
  bf16_t* attn = (bf16_t*)(ws + off); off += (size_t)B_ * P_ * H_ * 2; // 64 MB

  {
    int n = D_ * H_;
    k_transpose_cvt<<<(n + 255) / 256, 256, 0, stream>>>(W_w, WwT, D_, H_);
    n = H_ * H_;
    k_transpose_cvt<<<(n + 255) / 256, 256, 0, stream>>>(map_w, mapT, H_, H_);
  }

  dim3 blk(128);
  dim3 gP32(B_ * P_ / 32, H_ / 256);  // (1024, 4)
  dim3 gQ32(B_ * Q_ / 32, H_ / 256);

  // Wp = passage @ W_w + b ; Wq = question @ W_w + b (also Wq^T per batch)
  k_gemm_xw<<<gP32, blk, 0, stream>>>(passage, WwT, W_b, Wp, nullptr, D_, H_, P_);
  k_gemm_xw<<<gQ32, blk, 0, stream>>>(question, WwT, W_b, Wq, WqT, D_, H_, Q_);

  // scores + masked softmax -> alpha (bf16), fused
  k_scores_softmax<<<dim3(B_ * P_ / 16), 256, 0, stream>>>(Wp, Wq, q_mask, alp);

  // attn = alpha @ Wq  (B operand = Wq^T, per-batch stride H*Q)
  k_gemm_av<<<gP32, blk, 0, stream>>>(alp, WqT, nullptr, attn, nullptr, Q_, H_,
                                      (long)H_ * Q_, P_, 0);
  // out = relu(attn @ map_w + map_b) -> f32
  k_gemm_av<<<gP32, blk, 0, stream>>>(attn, mapT, map_b, nullptr, out, H_, H_, 0,
                                      P_, 1);
}
